// RQ_39135742001406
// MI455X (gfx1250) — matrix operand order repaired and checked by reading, not compiled, not run on hardware
//
#include <hip/hip_runtime.h>

#define S_ 4
#define K_ 1024
#define D_ 256
#define NT 65536
#define BETA_ 0.25f
#define DECAY_ 0.99f
#define EPS_ 1e-5f

// LDS static layout (dynamic smem base offset 0)
#define LDS_RES_OFF  0
#define LDS_B0_OFF   65536
#define LDS_B1_OFF   73728
#define LDS_IDX_OFF  81920
#define LDS_TOTAL    82176

typedef __attribute__((ext_vector_type(16))) __bf16 v16bf;
typedef __attribute__((ext_vector_type(8)))  float  v8f;

union V16A { unsigned int u[8]; v16bf v; };

__device__ __forceinline__ unsigned int f2bf_pk(float lo, float hi) {
    unsigned int ul = __float_as_uint(lo);
    ul = (ul + 0x7FFFu + ((ul >> 16) & 1u)) >> 16;
    unsigned int uh = __float_as_uint(hi);
    uh = (uh + 0x7FFFu + ((uh >> 16) & 1u)) >> 16;
    return (ul & 0xFFFFu) | (uh << 16);
}

// CDNA5 async global->LDS copy (ASYNCcnt), no VGPR round-trip.
__device__ __forceinline__ void async_g2l_b128(unsigned ldsOff, const void* gaddr) {
    asm volatile("global_load_async_to_lds_b128 %0, %1, off"
                 :: "v"(ldsOff), "v"((unsigned long long)(uintptr_t)gaddr)
                 : "memory");
}
#define WAIT_ASYNCCNT(n) asm volatile("s_wait_asynccnt " #n ::: "memory")

// ---- prep: codebooks fp32 -> bf16, c2[k] = ||C_k||^2, zero counts ----
__global__ void rvq_prep_cb(const float* __restrict__ cb,
                            unsigned short* __restrict__ cbbf,
                            float* __restrict__ c2,
                            float* __restrict__ cnt) {
    int t = blockIdx.x * blockDim.x + threadIdx.x;
    if (t >= S_ * K_) return;
    const float* src = cb + (size_t)t * D_;
    unsigned short* dst = cbbf + (size_t)t * D_;
    float s = 0.f;
    for (int d = 0; d < D_; ++d) {
        float f = src[d];
        s += f * f;
        unsigned int u = __float_as_uint(f);
        dst[d] = (unsigned short)((u + 0x7FFFu + ((u >> 16) & 1u)) >> 16);
    }
    c2[t] = s;
    cnt[t] = 0.f;
}

__global__ void rvq_zero(float* __restrict__ zsum, float* __restrict__ lossp) {
    int t = blockIdx.x * blockDim.x + threadIdx.x;
    if (t < S_ * K_ * D_) zsum[t] = 0.f;
    if (t == 0) *lossp = 0.f;
}

// ---- main fused kernel: 4 stages of WMMA distance + argmin + residual update ----
__global__ __launch_bounds__(128)
void rvq_main(const float* __restrict__ z, const float* __restrict__ cb,
              const unsigned short* __restrict__ cbbf, const float* __restrict__ c2,
              float* __restrict__ cnt, float* __restrict__ zsum,
              float* __restrict__ outq, float* __restrict__ lossp,
              int* __restrict__ idxOut) {
    extern __shared__ char smem[];
    float*  res  = (float*)(smem + LDS_RES_OFF);   // 64*256 fp32 (64 KB)
    __bf16* bt0  = (__bf16*)(smem + LDS_B0_OFF);   // 16*256 bf16 (8 KB)
    __bf16* bt1  = (__bf16*)(smem + LDS_B1_OFF);   // 16*256 bf16 (8 KB)
    int*    sIdx = (int*)(smem + LDS_IDX_OFF);     // 64 ints

    const int tid  = threadIdx.x;
    const int lane = tid & 31;
    const int wav  = tid >> 5;
    const int blockRow = blockIdx.x * 64;

    // stage-in: 64 rows of z into LDS residual
    for (int e = tid; e < 64 * 256; e += 128)
        res[e] = z[(size_t)blockRow * 256 + e];
    __syncthreads();

    const int m    = lane & 15;   // row (A) / column (B,C) index within tile
    const int half = lane >> 4;
    float sse = 0.f;

    for (int s = 0; s < S_; ++s) {
        // ---- preload this wave's 16x256 residual tile as bf16 A-fragments ----
        V16A a[8];
        const float* rrow = res + (wav * 16 + m) * 256;
        #pragma unroll
        for (int c = 0; c < 8; ++c) {
            int b0 = c * 32 + half * 16;      // K = half*16 .. +7    (VGPR 0..3)
            int b1 = c * 32 + half * 16 + 8;  // K = half*16+8 .. +7  (VGPR 4..7)
            #pragma unroll
            for (int v = 0; v < 4; ++v) {
                a[c].u[v]     = f2bf_pk(rrow[b0 + 2 * v], rrow[b0 + 2 * v + 1]);
                a[c].u[4 + v] = f2bf_pk(rrow[b1 + 2 * v], rrow[b1 + 2 * v + 1]);
            }
        }

        float mind[8];
        int   midx[8];
        #pragma unroll
        for (int j = 0; j < 8; ++j) { mind[j] = 3.4e38f; midx[j] = 0; }

        const unsigned short* cbs = cbbf + (size_t)s * K_ * D_;

        // prologue: async-stage tile kt=0 into buffer 0 (4x16B per thread)
        {
            const char* src0 = (const char*)cbs;
            #pragma unroll
            for (int i = 0; i < 4; ++i) {
                unsigned off = (unsigned)((tid + i * 128) * 16);
                async_g2l_b128(LDS_B0_OFF + off, src0 + off);
            }
        }

        for (int kt = 0; kt < K_ / 16; ++kt) {
            // issue async copy of tile kt+1 into the other buffer
            if (kt + 1 < K_ / 16) {
                const char* srcn = (const char*)(cbs + (size_t)(kt + 1) * 16 * D_);
                unsigned dbase = ((kt + 1) & 1) ? LDS_B1_OFF : LDS_B0_OFF;
                #pragma unroll
                for (int i = 0; i < 4; ++i) {
                    unsigned off = (unsigned)((tid + i * 128) * 16);
                    async_g2l_b128(dbase + off, srcn + off);
                }
                WAIT_ASYNCCNT(4);   // my 4 copies for tile kt (oldest) are done
            } else {
                WAIT_ASYNCCNT(0);
            }
            __syncthreads();        // all waves' tile-kt data visible in LDS

            const __bf16* btc = (kt & 1) ? bt1 : bt0;
            v8f acc = {};
            #pragma unroll
            for (int c = 0; c < 8; ++c) {
                // B layout: lane -> column n=m, K = c*32 + half*16 .. +15 (contiguous)
                const v16bf bv = *(const v16bf*)(btc + m * 256 + c * 32 + half * 16);
                acc = __builtin_amdgcn_wmma_f32_16x16x32_bf16(
                          false, a[c].v, false, bv, (short)0, acc, false, false);
            }

            int   col = kt * 16 + m;
            float c2v = c2[s * K_ + col];
            #pragma unroll
            for (int j = 0; j < 8; ++j) {
                float dd = __fmaf_rn(-2.f, acc[j], c2v);  // ||c||^2 - 2 r.c
                if (dd < mind[j]) { mind[j] = dd; midx[j] = col; }
            }
            __syncthreads();  // reads of buf[kt&1] done before it is re-targeted
        }

        // cross-lane argmin within each 16-lane half (lanes share the 8 rows)
        #pragma unroll
        for (int mask = 1; mask <= 8; mask <<= 1) {
            #pragma unroll
            for (int j = 0; j < 8; ++j) {
                float od = __shfl_xor(mind[j], mask, 32);
                int   oi = __shfl_xor(midx[j], mask, 32);
                if (od < mind[j] || (od == mind[j] && oi < midx[j])) {
                    mind[j] = od; midx[j] = oi;
                }
            }
        }
        if ((lane & 15) == 0) {
            #pragma unroll
            for (int j = 0; j < 8; ++j)
                sIdx[wav * 16 + half * 8 + j] = midx[j];
        }
        __syncthreads();

        // ---- update: residual -= C[idx] (fp32), loss SSE, EMA scatter ----
        for (int e = tid; e < 64 * 256; e += 128) {
            int row = e >> 8, d = e & 255;
            int id  = sIdx[row];
            float cv = cb[((size_t)s * K_ + id) * D_ + d];
            float r  = res[e];
            float df = cv - r;
            sse += df * df;
            atomicAdd(&zsum[((size_t)s * K_ + id) * D_ + d], r);
            res[e] = r - cv;
        }
        if (tid < 64) {
            atomicAdd(&cnt[s * K_ + sIdx[tid]], 1.f);
            idxOut[(size_t)s * NT + blockRow + tid] = sIdx[tid];
        }
        __syncthreads();
    }

    // z_q_out = recon = z - residual_final
    for (int e = tid; e < 64 * 256; e += 128)
        outq[(size_t)blockRow * 256 + e] = z[(size_t)blockRow * 256 + e] - res[e];

    // loss: wave reduce then one atomic per wave
    #pragma unroll
    for (int mask = 1; mask < 32; mask <<= 1)
        sse += __shfl_xor(sse, mask, 32);
    if (lane == 0)
        atomicAdd(lossp, sse * (BETA_ / ((float)NT * (float)D_)));
}

// ---- finalize: EMA codebook math ----
__global__ void rvq_finalize(const float* __restrict__ Nin, const float* __restrict__ zain,
                             const float* __restrict__ cnt, const float* __restrict__ zsum,
                             float* __restrict__ cbOut, float* __restrict__ NOut,
                             float* __restrict__ zaOut) {
    int t = blockIdx.x * blockDim.x + threadIdx.x;
    if (t >= S_ * K_ * D_) return;
    int sk = t >> 8;   // t / D_
    int d  = t & 255;
    float nk = cnt[sk];
    float Ns = DECAY_ * Nin[sk] + (1.f - DECAY_) * nk;
    float zk = zsum[t] / (nk + EPS_);
    float za = DECAY_ * zain[t] + (1.f - DECAY_) * zk;
    zaOut[t] = za;
    cbOut[t] = za / (Ns + EPS_);
    if (d == 0) NOut[sk] = Ns;
}

extern "C" void kernel_launch(void* const* d_in, const int* in_sizes, int n_in,
                              void* d_out, int out_size, void* d_ws, size_t ws_size,
                              hipStream_t stream) {
    const float* z    = (const float*)d_in[0];  // [64,1024,256]
    const float* cb   = (const float*)d_in[1];  // [4,1024,256]
    const float* Nin  = (const float*)d_in[2];  // [4,1024]
    const float* zain = (const float*)d_in[3];  // [4,1024,256]

    float* outF  = (float*)d_out;
    float* outQ  = outF;                                 // 16,777,216
    float* lossP = outF + 16777216;                      // 1
    int*   idxO  = (int*)(outF + 16777217);              // 262,144 (int32)
    float* cbO   = outF + 16777217 + 262144;             // 1,048,576
    float* NO    = cbO + 1048576;                        // 4,096
    float* zaO   = NO + 4096;                            // 1,048,576

    char* ws = (char*)d_ws;
    unsigned short* cbbf = (unsigned short*)ws;          // 2,097,152 B
    float* c2   = (float*)(ws + 2097152);                // 16,384 B
    float* cnt  = (float*)(ws + 2097152 + 16384);        // 16,384 B
    float* zsum = (float*)(ws + 2097152 + 32768);        // 4,194,304 B

    rvq_prep_cb<<<(S_ * K_ + 255) / 256, 256, 0, stream>>>(cb, cbbf, c2, cnt);
    rvq_zero<<<(S_ * K_ * D_ + 255) / 256, 256, 0, stream>>>(zsum, lossP);

    rvq_main<<<NT / 64, 128, LDS_TOTAL, stream>>>(z, cb, cbbf, c2, cnt, zsum,
                                                  outQ, lossP, idxO);

    rvq_finalize<<<(S_ * K_ * D_ + 255) / 256, 256, 0, stream>>>(Nin, zain, cnt, zsum,
                                                                 cbO, NO, zaO);
}
